// PointnetSAModuleMSG_19069654794722
// MI455X (gfx1250) — matrix operand order repaired and checked by reading, not compiled, not run on hardware
//
#include <hip/hip_runtime.h>
#include <hip/hip_bf16.h>
#include <cstdint>

typedef __attribute__((ext_vector_type(16))) __bf16 v16bf;
typedef __attribute__((ext_vector_type(8)))  float  v8f;

#define NB      8
#define NN      4096
#define NPT     1024
#define CFEAT   64
#define TOTC    384
#define EPSV    1e-5f

#define TILE_M  128
#define TILE_N  128
#define TILE_K  32
#define LDSP    (TILE_K + 8)   // row stride in bf16 elems (80 bytes = 5*16, keeps 16B alignment)

// ---------------- workspace layout (bytes) ----------------
static const size_t OFF_IDX  = 0;               // 262144*4 = 1 MiB max
static const size_t OFF_W    = 2u << 20;        // packed bf16 weights, transposed (<200 KB)
static const size_t OFF_STAT = 3u << 20;        // 2*256 f32 per layer
static const size_t OFF_X    = 4u << 20;        // bf16 activations: 262144*128*2 = 64 MiB
static const size_t OFF_Y    = (size_t)(4u + 64u) << 20; // f32 GEMM out: 262144*256*4 = 256 MiB

union FragBF { v16bf v; unsigned u[8]; };

// ---------------- trivial output head: new_xyz ----------------
__global__ void copy_newxyz(const float* __restrict__ xyz, float* __restrict__ out) {
    int i = blockIdx.x * 256 + threadIdx.x;
    if (i >= NB * NPT * 3) return;
    int c = i % 3; int bp = i / 3; int p = bp % NPT; int b = bp / NPT;
    out[i] = xyz[((size_t)b * NN + p) * 3 + c];
}

// ---------------- pack f32 weights -> bf16, TRANSPOSED (N-major), K and N padded ----------------
// wp[n * kpad + k] = w[k * cout + n]   (zero outside cin x cout)
__global__ void prep_w(const float* __restrict__ w, __bf16* __restrict__ wp,
                       int cin, int cout, int kpad, int npad) {
    int i = blockIdx.x * 256 + threadIdx.x;
    if (i >= npad * kpad) return;
    int k = i % kpad; int n = i / kpad;
    wp[i] = (k < cin && n < cout) ? (__bf16)w[(size_t)k * cout + n] : (__bf16)0.0f;
}

// ---------------- ball query (reference searchsorted semantics) ----------------
__global__ void ball_query(const float* __restrict__ xyz, float r2, int ns,
                           int* __restrict__ idx) {
    int t = blockIdx.x * 256 + threadIdx.x;
    if (t >= NB * NPT) return;
    int b = t / NPT, p = t % NPT;
    const float* q = xyz + ((size_t)b * NN + p) * 3;
    float qx = q[0], qy = q[1], qz = q[2];
    const float* base = xyz + (size_t)b * NN * 3;
    int cnt = 0, first = -1;
    for (int j = 0; j < NN; ++j) {
        float dx = base[j * 3 + 0] - qx;
        float dy = base[j * 3 + 1] - qy;
        float dz = base[j * 3 + 2] - qz;
        float d2 = dx * dx + dy * dy + dz * dz;
        if (d2 < r2) {
            if (cnt == 0) first = j;
            idx[(size_t)t * ns + cnt] = j;
            ++cnt;
            if (cnt >= ns) break;
        }
    }
    int pad = (cnt == 0) ? (NN - 1) : first;
    for (int k = cnt; k < ns; ++k) idx[(size_t)t * ns + k] = pad;
}

// ---------------- gather + center + concat -> bf16 rows x kpad ----------------
__global__ void gather_concat(const float* __restrict__ xyz, const float* __restrict__ feat,
                              const int* __restrict__ idx, __bf16* __restrict__ X,
                              int ns, int kpad) {
    int r = blockIdx.x * 256 + threadIdx.x;
    int rows = NB * NPT * ns;
    if (r >= rows) return;
    int bp = r / ns; int p = bp % NPT; int b = bp / NPT;
    int j = idx[r];
    const float* pj = xyz + ((size_t)b * NN + j) * 3;
    const float* pq = xyz + ((size_t)b * NN + p) * 3;
    __bf16* dst = X + (size_t)r * kpad;
    dst[0] = (__bf16)(pj[0] - pq[0]);
    dst[1] = (__bf16)(pj[1] - pq[1]);
    dst[2] = (__bf16)(pj[2] - pq[2]);
    const float* f = feat + ((size_t)b * NN + j) * CFEAT;
#pragma unroll 8
    for (int c = 0; c < CFEAT; ++c) dst[3 + c] = (__bf16)f[c];
    for (int c = 3 + CFEAT; c < kpad; ++c) dst[c] = (__bf16)0.0f;
}

// ---------------- WMMA GEMM + BN statistics epilogue ----------------
// Y[M x N] = A[M x K](bf16) * BwT[Npad x K](bf16, pre-transposed), f32 accum.
// 128 threads = 4 waves; each wave computes a 32x128 strip (2 A-frags x 8 B-frags = 16 WMMA/K-step).
// Tiles staged with GLOBAL_LOAD_ASYNC_TO_LDS_B128 (ASYNCcnt), double-buffered.
__global__ __launch_bounds__(128) void gemm_bn_stats(
    const __bf16* __restrict__ A, const __bf16* __restrict__ BwT,
    float* __restrict__ Y, int K, int N, float* __restrict__ stat) {

    __shared__ __align__(16) __bf16 As [2][TILE_M][LDSP];  // A tile, K contiguous per row
    __shared__ __align__(16) __bf16 BsT[2][TILE_N][LDSP];  // B tile, K contiguous per column
    __shared__ float sSum[TILE_N], sSsq[TILE_N];

    const int tid  = threadIdx.x;
    const int wave = tid >> 5;          // 0..3
    const int lane = tid & 31;
    const int half = lane >> 4;
    const int l16  = lane & 15;
    const int m0   = blockIdx.x * TILE_M;
    const int n0   = blockIdx.y * TILE_N;

    v8f acc[2][8];
#pragma unroll
    for (int mi = 0; mi < 2; ++mi)
#pragma unroll
        for (int j = 0; j < 8; ++j)
#pragma unroll
            for (int r = 0; r < 8; ++r) acc[mi][j][r] = 0.0f;

    // one 64-byte K-row per thread for each of A and B^T (128 rows each)
    const __bf16* gA = A   + (size_t)(m0 + tid) * K;
    const __bf16* gB = BwT + (size_t)(n0 + tid) * K;   // Npad >= n0+128 guaranteed by packing

    auto issue = [&](int kk, int b) {
        unsigned la = (unsigned)(size_t)&As[b][tid][0];
        const __bf16* pa = gA + kk;
        asm volatile(
            "global_load_async_to_lds_b128 %0, %1, off\n\t"
            "global_load_async_to_lds_b128 %0, %1, off offset:16\n\t"
            "global_load_async_to_lds_b128 %0, %1, off offset:32\n\t"
            "global_load_async_to_lds_b128 %0, %1, off offset:48"
            :: "v"(la), "v"(pa) : "memory");
        unsigned lb = (unsigned)(size_t)&BsT[b][tid][0];
        const __bf16* pb = gB + kk;
        asm volatile(
            "global_load_async_to_lds_b128 %0, %1, off\n\t"
            "global_load_async_to_lds_b128 %0, %1, off offset:16\n\t"
            "global_load_async_to_lds_b128 %0, %1, off offset:32\n\t"
            "global_load_async_to_lds_b128 %0, %1, off offset:48"
            :: "v"(lb), "v"(pb) : "memory");
    };

    issue(0, 0);
    int buf = 0;
    for (int k0 = 0; k0 < K; k0 += TILE_K) {
        asm volatile("s_wait_asynccnt 0x0" ::: "memory");
        __syncthreads();                       // tile(i) visible; compute(i-1) finished everywhere
        if (k0 + TILE_K < K) issue(k0 + TILE_K, buf ^ 1);   // overlap next fetch with compute

        // A fragments (ISA 16-bit A 16x32 layout: v<4 -> K=2v(+8*half), v>=4 -> K=16+2(v-4)(+8*half))
        FragBF fa0, fa1;
        {
            int row = wave * 32 + l16;
#pragma unroll
            for (int v = 0; v < 8; ++v) {
                int kb = (v < 4) ? (2 * v + 8 * half) : (16 + 2 * (v - 4) + 8 * half);
                fa0.u[v] = *(const unsigned*)&As[buf][row][kb];
                fa1.u[v] = *(const unsigned*)&As[buf][row + 16][kb];
            }
        }
#pragma unroll
        for (int j = 0; j < 8; ++j) {
            // B fragment (same K order as A: v<4 -> K=2v(+8*half), v>=4 -> K=16+2(v-4)(+8*half); N = l16)
            FragBF fb;
            int col = j * 16 + l16;
#pragma unroll
            for (int v = 0; v < 8; ++v)
                fb.u[v] = *(const unsigned*)&BsT[buf][col][(v < 4) ? (2 * v + 8 * half) : (16 + 2 * (v - 4) + 8 * half)];
            acc[0][j] = __builtin_amdgcn_wmma_f32_16x16x32_bf16(
                false, fa0.v, false, fb.v, (short)0, acc[0][j], false, false);
            acc[1][j] = __builtin_amdgcn_wmma_f32_16x16x32_bf16(
                false, fa1.v, false, fb.v, (short)0, acc[1][j], false, false);
        }
        buf ^= 1;
    }

    // Epilogue: store Y, per-column sum / sumsq (regs -> LDS -> global f32 atomics)
    if (tid < TILE_N) { sSum[tid] = 0.0f; sSsq[tid] = 0.0f; }
    __syncthreads();
#pragma unroll
    for (int mi = 0; mi < 2; ++mi)
#pragma unroll
        for (int j = 0; j < 8; ++j) {
            int col = n0 + j * 16 + l16;
            float ls = 0.0f, lq = 0.0f;
#pragma unroll
            for (int r = 0; r < 8; ++r) {
                int row = m0 + wave * 32 + mi * 16 + r + 8 * half;
                float y = acc[mi][j][r];
                if (col < N) Y[(size_t)row * N + col] = y;
                ls += y; lq += y * y;
            }
            if (col < N) {
                atomicAdd(&sSum[j * 16 + l16], ls);
                atomicAdd(&sSsq[j * 16 + l16], lq);
            }
        }
    __syncthreads();
    if (tid < TILE_N) {
        int col = n0 + tid;
        if (col < N) {
            atomicAdd(&stat[col],     sSum[tid]);
            atomicAdd(&stat[N + col], sSsq[tid]);
        }
    }
}

// ---------------- BN + ReLU -> bf16 next-layer input ----------------
__global__ void bn_relu_to_bf16(const float* __restrict__ Y, const float* __restrict__ stat,
                                const float* __restrict__ g, const float* __restrict__ bt,
                                __bf16* __restrict__ Xo, int total, int N, float invM) {
    int i = blockIdx.x * 256 + threadIdx.x;
    if (i >= total) return;
    int col = i % N;
    float mean = stat[col] * invM;
    float var  = stat[N + col] * invM - mean * mean;
    float sc   = __frsqrt_rn(var + EPSV) * g[col];
    float xh   = (Y[i] - mean) * sc + bt[col];
    Xo[i] = (__bf16)fmaxf(xh, 0.0f);
}

// ---------------- last layer: BN + ReLU + max over samples, transposed store ----------------
__global__ void bn_relu_max_out(const float* __restrict__ Y, const float* __restrict__ stat,
                                const float* __restrict__ g, const float* __restrict__ bt,
                                float* __restrict__ outF, int ns, int N, int chOff, float invM) {
    int i = blockIdx.x * 256 + threadIdx.x;
    if (i >= NB * NPT * N) return;
    int c = i % N; int bp = i / N; int p = bp % NPT; int b = bp / NPT;
    float mean = stat[c] * invM;
    float var  = stat[N + c] * invM - mean * mean;
    float sc   = __frsqrt_rn(var + EPSV) * g[c];
    float sh   = bt[c] - mean * sc;
    const float* base = Y + (size_t)bp * ns * N + c;
    float m = 0.0f;   // ReLU outputs are >= 0
    for (int s = 0; s < ns; ++s) {
        float xh = fmaxf(base[(size_t)s * N] * sc + sh, 0.0f);
        m = fmaxf(m, xh);
    }
    outF[((size_t)b * TOTC + chOff + c) * NPT + p] = m;
}

// ---------------- host ----------------
struct LayerCfg { int widx; int cin; int cout; int kpad; int npad; size_t woff; };

extern "C" void kernel_launch(void* const* d_in, const int* in_sizes, int n_in,
                              void* d_out, int out_size, void* d_ws, size_t ws_size,
                              hipStream_t stream) {
    (void)in_sizes; (void)n_in; (void)out_size; (void)ws_size;
    const float* xyz  = (const float*)d_in[0];
    const float* feat = (const float*)d_in[1];
    float* out = (float*)d_out;

    char*   ws   = (char*)d_ws;
    int*    idx  = (int*)   (ws + OFF_IDX);
    __bf16* Wp   = (__bf16*)(ws + OFF_W);
    float*  stat = (float*) (ws + OFF_STAT);
    __bf16* X    = (__bf16*)(ws + OFF_X);
    float*  Y    = (float*) (ws + OFF_Y);

    // output head: new_xyz
    copy_newxyz<<<(NB * NPT * 3 + 255) / 256, 256, 0, stream>>>(xyz, out);

    // layer configs; weight slot offsets in bf16 elements (transposed, npad x kpad)
    static const LayerCfg layers[2][3] = {
        { {2, 67,  64,  96, 128, 0},
          {5, 64,  64,  64, 128, 12288},
          {8, 64, 128,  64, 128, 12288 + 8192} },
        { {11, 67, 128,  96, 128, 12288 + 8192 + 8192},
          {14,128, 128, 128, 128, 12288 + 8192 + 8192 + 12288},
          {17,128, 256, 128, 256, 12288 + 8192 + 8192 + 12288 + 16384} }
    };

    // pack all weights once (transposed + padded)
    for (int br = 0; br < 2; ++br)
        for (int l = 0; l < 3; ++l) {
            const LayerCfg& L = layers[br][l];
            int n = L.npad * L.kpad;
            prep_w<<<(n + 255) / 256, 256, 0, stream>>>(
                (const float*)d_in[L.widx], Wp + L.woff, L.cin, L.cout, L.kpad, L.npad);
        }

    const float radii2[2] = {0.2f * 0.2f, 0.4f * 0.4f};
    const int   nss[2]    = {16, 32};
    const int   choff[2]  = {0, 128};

    for (int br = 0; br < 2; ++br) {
        int ns   = nss[br];
        int rows = NB * NPT * ns;            // 131072 / 262144, multiples of 128

        ball_query<<<(NB * NPT + 255) / 256, 256, 0, stream>>>(xyz, radii2[br], ns, idx);
        gather_concat<<<(rows + 255) / 256, 256, 0, stream>>>(xyz, feat, idx, X, ns, 96);

        for (int l = 0; l < 3; ++l) {
            const LayerCfg& L = layers[br][l];
            const float* g  = (const float*)d_in[L.widx + 1];
            const float* bt = (const float*)d_in[L.widx + 2];

            hipMemsetAsync(stat, 0, (size_t)2 * L.cout * sizeof(float), stream);

            dim3 grid(rows / TILE_M, (L.cout + TILE_N - 1) / TILE_N);
            gemm_bn_stats<<<grid, 128, 0, stream>>>(X, Wp + L.woff, Y, L.kpad, L.cout, stat);

            float invM = 1.0f / (float)rows;
            if (l < 2) {
                int n = rows * L.cout;
                bn_relu_to_bf16<<<(n + 255) / 256, 256, 0, stream>>>(
                    Y, stat, g, bt, X, n, L.cout, invM);
            } else {
                int n = NB * NPT * L.cout;
                bn_relu_max_out<<<(n + 255) / 256, 256, 0, stream>>>(
                    Y, stat, g, bt, out + NB * NPT * 3, ns, L.cout, choff[br], invM);
            }
        }
    }
}
